// DecoderOnlyAttention_68264210202844
// MI455X (gfx1250) — matrix operand order repaired and checked by reading, not compiled, not run on hardware
//
#include <hip/hip_runtime.h>

// ---------------------------------------------------------------------------
// Decoder-only attention prefill for MI455X (gfx1250), bf16 WMMA everywhere.
// B=1, S=2048, HID=4096, H=32, KVH=8, D=128, KV_LEN=2048, seq_pos=0.
// Round 3: Q tile in flash-attention loaded via TDM (tensor_load_to_lds,
//          TENSORcnt); GEMM staging stays on GLOBAL_LOAD_ASYNC_TO_LDS_B128.
// ---------------------------------------------------------------------------

typedef __attribute__((ext_vector_type(16))) __bf16 v16bf;
typedef __attribute__((ext_vector_type(8)))  float  v8f;
typedef __attribute__((ext_vector_type(4)))  unsigned int u32x4;
typedef __attribute__((ext_vector_type(8)))  unsigned int u32x8;

#define WMMA_BF16(a, b, c) \
  __builtin_amdgcn_wmma_f32_16x16x32_bf16(false, (a), false, (b), (short)0, (c), false, false)

union Frag16 { v16bf v; uint4 u[2]; };

static __device__ __forceinline__ unsigned short f2bf(float f) {
  union { float f; unsigned int u; } c; c.f = f;
  unsigned int r = c.u + 0x7FFFu + ((c.u >> 16) & 1u);  // round-to-nearest-even
  return (unsigned short)(r >> 16);
}

// Generic LDS pointer -> raw LDS byte offset (low 32 bits of the address).
static __device__ __forceinline__ unsigned lds_off(const void* p) {
  return (unsigned)(unsigned long long)p;
}

// 16B async copy global -> LDS, tracked by ASYNCcnt.
static __device__ __forceinline__ void async_ld_b128(const void* gaddr, unsigned laddr) {
  asm volatile("global_load_async_to_lds_b128 %0, %1, off"
               :: "v"(laddr), "v"(gaddr)
               : "memory");
}

static __device__ __forceinline__ void wait_async0() {
#if __has_builtin(__builtin_amdgcn_s_wait_asynccnt)
  __builtin_amdgcn_s_wait_asynccnt(0);
#else
  asm volatile("s_wait_asynccnt 0x0" ::: "memory");
#endif
}

static __device__ __forceinline__ void wait_tensor0() {
#if __has_builtin(__builtin_amdgcn_s_wait_tensorcnt)
  __builtin_amdgcn_s_wait_tensorcnt(0);
#else
  asm volatile("s_wait_tensorcnt 0x0" ::: "memory");
#endif
}

// TDM: 2D tile global->LDS via Tensor DMA descriptor (2 SGPR groups,
// VADDR2/VADDR3 = NULL => tensors up to 2D). Tracked by TENSORcnt.
// rows x cols elements of 2 bytes, global row stride = cols (packed tile).
static __device__ __forceinline__ void tdm_load_2d_bf16(const void* gaddr, unsigned laddr,
                                                        unsigned rows, unsigned cols,
                                                        unsigned row_stride_elems) {
  unsigned long long ga = (unsigned long long)gaddr;
  u32x4 g0;
  g0[0] = 1u;                                           // count=1, user mode
  g0[1] = laddr;                                        // lds_addr
  g0[2] = (unsigned)ga;                                 // global_addr[31:0]
  g0[3] = (unsigned)((ga >> 32) & 0x01FFFFFFu) | 0x80000000u;  // ga[56:32] | type=2
  u32x8 g1;
  g1[0] = 1u << 16;                                     // data_size=1 (2B), mask=0
  g1[1] = (cols & 0xFFFFu) << 16;                       // tensor_dim0[15:0] @bit48
  g1[2] = ((cols >> 16) & 0xFFFFu) | ((rows & 0xFFFFu) << 16);  // dim0 hi | tensor_dim1 lo
  g1[3] = ((rows >> 16) & 0xFFFFu) | ((cols & 0xFFFFu) << 16);  // dim1 hi | tile_dim0
  g1[4] = rows & 0xFFFFu;                               // tile_dim1 (tile_dim2=0)
  g1[5] = row_stride_elems;                             // tensor_dim0_stride[31:0]
  g1[6] = 0u;                                           // stride hi | dim1_stride lo
  g1[7] = 0u;
  asm volatile("tensor_load_to_lds %0, %1"
               :: "s"(g0), "s"(g1)
               : "memory");
}

// ---------------------------------------------------------------------------
// elementwise f32 -> bf16
// ---------------------------------------------------------------------------
__global__ __launch_bounds__(256) void cast_f32_bf16(const float* __restrict__ x,
                                                     unsigned short* __restrict__ y,
                                                     int n) {
  int i = blockIdx.x * 256 + threadIdx.x;
  if (i < n) y[i] = f2bf(x[i]);
}

// ---------------------------------------------------------------------------
// W [Nrows][Kcols] f32  ->  WT [Kcols][Nrows] bf16   (32x32 LDS tile transpose)
// ---------------------------------------------------------------------------
__global__ __launch_bounds__(256) void cast_transpose(const float* __restrict__ W,
                                                      unsigned short* __restrict__ WT,
                                                      int Nrows, int Kcols) {
  __shared__ float tile[32][33];
  int tx = threadIdx.x & 31, ty = threadIdx.x >> 5;  // 32 x 8
  int bx = blockIdx.x;  // K tile
  int by = blockIdx.y;  // N tile
#pragma unroll
  for (int i = 0; i < 32; i += 8)
    tile[ty + i][tx] = W[(size_t)(by * 32 + ty + i) * Kcols + bx * 32 + tx];
  __syncthreads();
#pragma unroll
  for (int i = 0; i < 32; i += 8) {
    int k = bx * 32 + ty + i;
    int n = by * 32 + tx;
    WT[(size_t)k * Nrows + n] = f2bf(tile[tx][ty + i]);
  }
}

// ---------------------------------------------------------------------------
// C[M][N] f32 = A[M][K] bf16 x BT[K][N] bf16.
// 128x128 block tile, KT=32, 8 waves (4 M x 2 N), wave tile 32x64 = 2x4 WMMA.
// A/B tiles staged with async global->LDS copies.
// ---------------------------------------------------------------------------
__global__ __launch_bounds__(256) void gemm_bf16(const unsigned short* __restrict__ A,
                                                 const unsigned short* __restrict__ BT,
                                                 float* __restrict__ C,
                                                 int M, int N, int K) {
  __shared__ unsigned short As[128 * 32];  // [m][k]
  __shared__ unsigned short Bs[32 * 128];  // [k][n]

  const int tid = threadIdx.x;
  const int lane = tid & 31, wave = tid >> 5;
  const int wm = wave & 3, wn = wave >> 2;       // 4 x 2 wave grid
  const int r = lane & 15, kg = lane >> 4;       // A-fragment row / K-half
  const int by = blockIdx.y, bx = blockIdx.x;

  v8f acc[2][4];
#pragma unroll
  for (int mt = 0; mt < 2; ++mt)
#pragma unroll
    for (int nt = 0; nt < 4; ++nt) acc[mt][nt] = (v8f){};

  const int arow = tid >> 1, aseg = tid & 1;     // A: 128 rows x 2 halves of 16
  const int brow = tid >> 3, bseg = tid & 7;     // B: 32 rows x 8 segs of 16

  const unsigned la = lds_off(As + arow * 32 + aseg * 16);
  const unsigned lb = lds_off(Bs + brow * 128 + bseg * 16);

  for (int k0 = 0; k0 < K; k0 += 32) {
    __syncthreads();  // previous tile fully consumed
    {
      const unsigned short* ga = A + (size_t)(by * 128 + arow) * K + k0 + aseg * 16;
      async_ld_b128(ga, la);
      async_ld_b128(ga + 8, la + 16);
      const unsigned short* gb = BT + (size_t)(k0 + brow) * N + bx * 128 + bseg * 16;
      async_ld_b128(gb, lb);
      async_ld_b128(gb + 8, lb + 16);
    }
    if (k0 + 32 < K) {  // prefetch next tiles -> global_prefetch_b8
      __builtin_prefetch(A + (size_t)(by * 128 + arow) * K + k0 + 32 + aseg * 16, 0, 1);
      __builtin_prefetch(BT + (size_t)(k0 + 32 + brow) * N + bx * 128 + bseg * 16, 0, 1);
    }
    wait_async0();
    __syncthreads();

    Frag16 a[2], b[4];
#pragma unroll
    for (int mt = 0; mt < 2; ++mt) {
      const unsigned short* p = As + (wm * 32 + mt * 16 + r) * 32;
      a[mt].u[0] = *(const uint4*)(p + kg * 8);        // K = kg*8 .. +7
      a[mt].u[1] = *(const uint4*)(p + 16 + kg * 8);   // K = 16+kg*8 .. +7
    }
#pragma unroll
    for (int nt = 0; nt < 4; ++nt) {
      const unsigned short* p = Bs + wn * 64 + nt * 16 + r;  // lane = N col, K in table order
#pragma unroll
      for (int i = 0; i < 8; ++i) {
        b[nt].v[i] = __builtin_bit_cast(__bf16, p[(kg * 8 + i) * 128]);
        b[nt].v[8 + i] = __builtin_bit_cast(__bf16, p[(16 + kg * 8 + i) * 128]);
      }
    }
#pragma unroll
    for (int mt = 0; mt < 2; ++mt)
#pragma unroll
      for (int nt = 0; nt < 4; ++nt)
        acc[mt][nt] = WMMA_BF16(a[mt].v, b[nt].v, acc[mt][nt]);
  }

  const int row0 = by * 128 + wm * 32;
  const int col0 = bx * 128 + wn * 64;
  const int hl = lane >> 4;
#pragma unroll
  for (int mt = 0; mt < 2; ++mt)
#pragma unroll
    for (int nt = 0; nt < 4; ++nt)
#pragma unroll
      for (int j = 0; j < 8; ++j) {
        int rr = row0 + mt * 16 + j + hl * 8;
        int cc = col0 + nt * 16 + (lane & 15);
        C[(size_t)rr * N + cc] = acc[mt][nt][j];
      }
}

// ---------------------------------------------------------------------------
// RoPE for Q: qf [S][H*128] f32 -> qb bf16 [H][S][128]
// ---------------------------------------------------------------------------
__global__ __launch_bounds__(256) void rope_q(const float* __restrict__ qf,
                                              const float* __restrict__ cosp,
                                              const float* __restrict__ sinp,
                                              unsigned short* __restrict__ qb) {
  int i = blockIdx.x * 256 + threadIdx.x;
  if (i >= 2048 * 32 * 128) return;
  int s = i >> 12;
  int rem = i & 4095;
  int h = rem >> 7;
  int d = rem & 127;
  float v = qf[i];
  float c = cosp[s * 128 + d], sn = sinp[s * 128 + d];
  float rot = (d < 64) ? -qf[i + 64] : qf[i - 64];
  float out = v * c + rot * sn;
  qb[((size_t)h * 2048 + s) * 128 + d] = f2bf(out);
}

// ---------------------------------------------------------------------------
// RoPE for K + copy V; writes fp32 caches (d_out) and bf16 head-major copies.
// kf/vf: [S][KVH*128] f32 ; caches/kb/vb: [KVH][2048][128]
// ---------------------------------------------------------------------------
__global__ __launch_bounds__(256) void rope_copy_kv(const float* __restrict__ kf,
                                                    const float* __restrict__ vf,
                                                    const float* __restrict__ cosp,
                                                    const float* __restrict__ sinp,
                                                    float* __restrict__ kcache,
                                                    float* __restrict__ vcache,
                                                    unsigned short* __restrict__ kb,
                                                    unsigned short* __restrict__ vb) {
  int i = blockIdx.x * 256 + threadIdx.x;
  if (i >= 2048 * 8 * 128) return;
  int s = i >> 10;
  int rem = i & 1023;
  int kvh = rem >> 7;
  int d = rem & 127;
  float c = cosp[s * 128 + d], sn = sinp[s * 128 + d];
  float kv = kf[i];
  float rot = (d < 64) ? -kf[i + 64] : kf[i - 64];
  float rk = kv * c + rot * sn;
  size_t ci = ((size_t)kvh * 2048 + s) * 128 + d;
  kcache[ci] = rk;
  kb[ci] = f2bf(rk);
  float vv = vf[i];
  vcache[ci] = vv;
  vb[ci] = f2bf(vv);
}

// ---------------------------------------------------------------------------
// Flash attention, causal, GQA (4 q-heads per kv-head).
// Block = (q-tile of 128 rows, head). 8 waves, each owns 16 q rows.
// Q tile: TDM tensor_load_to_lds (wave 0 issues, TENSORcnt).
// V tile: async global->LDS. K: VGPR transpose scatter.
// ---------------------------------------------------------------------------
__global__ __launch_bounds__(256) void flash_attn(const unsigned short* __restrict__ Qb,  // [H][2048][128]
                                                  const unsigned short* __restrict__ Kb,  // [KVH][2048][128]
                                                  const unsigned short* __restrict__ Vb,  // [KVH][2048][128]
                                                  unsigned short* __restrict__ Ctx) {     // [2048][4096]
  __shared__ unsigned short Qs[128 * 128];    // 32 KB  [q][d]
  __shared__ unsigned short KtPs[128 * 64];   // 16 KB  Kt: [d][kv] ; reused as per-wave P [16][64]
  __shared__ unsigned short Vs[64 * 128];     // 16 KB  [kv][d]

  const int qt = blockIdx.x;          // q tile (128 rows)
  const int h = blockIdx.y;
  const int kvh = h >> 2;
  const int tid = threadIdx.x;
  const int lane = tid & 31, wave = tid >> 5;
  const int r = lane & 15, kg = lane >> 4, hl = lane >> 4;
  const float scale = 0.08838834764831845f;   // 1/sqrt(128)

  if (wave == 0) {
    // TDM: one 128x128 bf16 tile, packed (row stride 128 elems), into Qs.
    tdm_load_2d_bf16(Qb + ((size_t)h * 2048 + qt * 128) * 128, lds_off(Qs),
                     /*rows=*/128, /*cols=*/128, /*row_stride=*/128);
    wait_tensor0();
  }

  float rowmax[8], rowsum[8];
  v8f o[8];
#pragma unroll
  for (int j = 0; j < 8; ++j) { rowmax[j] = -1e30f; rowsum[j] = 0.f; }
#pragma unroll
  for (int dt = 0; dt < 8; ++dt) o[dt] = (v8f){};

  const int nkb = (qt + 1) * 2;  // causal: kv blocks of 64 up to q-tile end

  for (int kb = 0; kb < nkb; ++kb) {
    __syncthreads();  // (also publishes TDM-loaded Q on first iteration)
    {  // async load V tile -> LDS [kv][d]
      int row = tid >> 2;
      int seg = (tid & 3) * 32;
      const unsigned short* src = Vb + ((size_t)kvh * 2048 + kb * 64 + row) * 128 + seg;
      unsigned dst = lds_off(Vs + row * 128 + seg);
#pragma unroll
      for (int i = 0; i < 4; ++i) async_ld_b128(src + i * 8, dst + i * 16);
    }
    {  // K tile -> LDS transposed [d][64] (VGPR scatter path)
      int row = tid >> 2;
      int dseg = (tid & 3) * 32;
      const uint4* src = (const uint4*)(Kb + ((size_t)kvh * 2048 + kb * 64 + row) * 128 + dseg);
#pragma unroll
      for (int i = 0; i < 4; ++i) {
        uint4 u = src[i];
        const unsigned short* e = (const unsigned short*)&u;
#pragma unroll
        for (int j = 0; j < 8; ++j) KtPs[(dseg + i * 8 + j) * 64 + row] = e[j];
      }
    }
    wait_async0();  // V staged
    __syncthreads();

    // ---- scores S = Q K^T : wave rows [wave*16, +16), 64 kv cols ----
    v8f s[4];
#pragma unroll
    for (int nt = 0; nt < 4; ++nt) s[nt] = (v8f){};
#pragma unroll
    for (int dk = 0; dk < 4; ++dk) {
      Frag16 a;
      const unsigned short* ap = Qs + (wave * 16 + r) * 128 + dk * 32;
      a.u[0] = *(const uint4*)(ap + kg * 8);
      a.u[1] = *(const uint4*)(ap + 16 + kg * 8);
#pragma unroll
      for (int nt = 0; nt < 4; ++nt) {
        Frag16 b;
        const unsigned short* bp = KtPs + (dk * 32) * 64 + nt * 16 + r;
#pragma unroll
        for (int i = 0; i < 8; ++i) {
          b.v[i] = __builtin_bit_cast(__bf16, bp[(kg * 8 + i) * 64]);
          b.v[8 + i] = __builtin_bit_cast(__bf16, bp[(16 + kg * 8 + i) * 64]);
        }
        s[nt] = WMMA_BF16(a.v, b.v, s[nt]);
      }
    }

    // ---- scale + causal mask ----
    const int rowg0 = qt * 128 + wave * 16 + hl * 8;  // + j
    const int col0 = kb * 64 + (lane & 15);
#pragma unroll
    for (int nt = 0; nt < 4; ++nt)
#pragma unroll
      for (int j = 0; j < 8; ++j) {
        float x = s[nt][j] * scale;
        if (col0 + nt * 16 > rowg0 + j) x = -1e9f;
        s[nt][j] = x;
      }

    // ---- online softmax (row stats live per lane-half) ----
    float bm[8];
#pragma unroll
    for (int j = 0; j < 8; ++j) {
      float m = s[0][j];
      m = fmaxf(m, s[1][j]); m = fmaxf(m, s[2][j]); m = fmaxf(m, s[3][j]);
      bm[j] = m;
    }
#pragma unroll
    for (int off = 1; off < 16; off <<= 1)
#pragma unroll
      for (int j = 0; j < 8; ++j) bm[j] = fmaxf(bm[j], __shfl_xor(bm[j], off, 32));

    float fac[8];
#pragma unroll
    for (int j = 0; j < 8; ++j) {
      float mn = fmaxf(rowmax[j], bm[j]);
      fac[j] = __expf(rowmax[j] - mn);
      rowmax[j] = mn;
    }

    float bs[8];
#pragma unroll
    for (int j = 0; j < 8; ++j) bs[j] = 0.f;
#pragma unroll
    for (int nt = 0; nt < 4; ++nt)
#pragma unroll
      for (int j = 0; j < 8; ++j) {
        float p = __expf(s[nt][j] - rowmax[j]);
        s[nt][j] = p;
        bs[j] += p;
      }
#pragma unroll
    for (int off = 1; off < 16; off <<= 1)
#pragma unroll
      for (int j = 0; j < 8; ++j) bs[j] += __shfl_xor(bs[j], off, 32);
#pragma unroll
    for (int j = 0; j < 8; ++j) rowsum[j] = rowsum[j] * fac[j] + bs[j];
#pragma unroll
    for (int dt = 0; dt < 8; ++dt)
#pragma unroll
      for (int j = 0; j < 8; ++j) o[dt][j] *= fac[j];

    __syncthreads();  // everyone done reading Kt before P overwrites it

    // ---- spill P (C layout) to per-wave LDS region, bf16 ----
    unsigned short* Ps = KtPs + wave * 1024;  // [16][64]
#pragma unroll
    for (int nt = 0; nt < 4; ++nt)
#pragma unroll
      for (int j = 0; j < 8; ++j)
        Ps[(j + hl * 8) * 64 + nt * 16 + (lane & 15)] = f2bf(s[nt][j]);

    // ---- O += P V ----
#pragma unroll
    for (int kc = 0; kc < 2; ++kc) {
      Frag16 a;
      const unsigned short* ap = Ps + r * 64 + kc * 32;
      a.u[0] = *(const uint4*)(ap + kg * 8);
      a.u[1] = *(const uint4*)(ap + 16 + kg * 8);
#pragma unroll
      for (int dt = 0; dt < 8; ++dt) {
        Frag16 b;
        const unsigned short* vp = Vs + (kc * 32) * 128 + dt * 16 + r;
#pragma unroll
        for (int i = 0; i < 8; ++i) {
          b.v[i] = __builtin_bit_cast(__bf16, vp[(kg * 8 + i) * 128]);
          b.v[8 + i] = __builtin_bit_cast(__bf16, vp[(16 + kg * 8 + i) * 128]);
        }
        o[dt] = WMMA_BF16(a.v, b.v, o[dt]);
      }
    }
  }

  // ---- epilogue: O /= l ; store ctx bf16 [s][h*128 + d] ----
#pragma unroll
  for (int dt = 0; dt < 8; ++dt)
#pragma unroll
    for (int j = 0; j < 8; ++j) {
      float val = o[dt][j] / rowsum[j];
      int rowl = wave * 16 + j + hl * 8;
      Ctx[(size_t)(qt * 128 + rowl) * 4096 + h * 128 + dt * 16 + (lane & 15)] = f2bf(val);
    }
}

// ---------------------------------------------------------------------------
// Host launcher
// ---------------------------------------------------------------------------
extern "C" void kernel_launch(void* const* d_in, const int* in_sizes, int n_in,
                              void* d_out, int out_size, void* d_ws, size_t ws_size,
                              hipStream_t stream) {
  const float* hs   = (const float*)d_in[0];
  const float* cosp = (const float*)d_in[2];
  const float* sinp = (const float*)d_in[3];
  const float* Wq   = (const float*)d_in[6];
  const float* Wk   = (const float*)d_in[7];
  const float* Wv   = (const float*)d_in[8];
  const float* Wo   = (const float*)d_in[9];
  // seq_positions == 0, past_* zeros, mask == tril -> fully causal overwrite.

  float* out    = (float*)d_out;
  float* attn   = out;                       // [2048][4096]
  float* kcache = out + 8388608;             // [8][2048][128]
  float* vcache = out + 10485760;            // [8][2048][128]

  char* w = (char*)d_ws;  // requires ~176 MB of scratch
  unsigned short* hsb  = (unsigned short*)(w + 0);          // 16 MB
  unsigned short* WqT  = (unsigned short*)(w + 16777216);   // 32 MB
  unsigned short* WkT  = (unsigned short*)(w + 50331648);   //  8 MB
  unsigned short* WvT  = (unsigned short*)(w + 58720256);   //  8 MB
  unsigned short* WoT  = (unsigned short*)(w + 67108864);   // 32 MB
  float*          qf   = (float*)(w + 100663296);           // 32 MB (reused as ctx)
  float*          kf   = (float*)(w + 134217728);           //  8 MB
  float*          vf   = (float*)(w + 142606336);           //  8 MB
  unsigned short* qb   = (unsigned short*)(w + 150994944);  // 16 MB
  unsigned short* kb   = (unsigned short*)(w + 167772160);  //  4 MB
  unsigned short* vb   = (unsigned short*)(w + 171966464);  //  4 MB
  unsigned short* ctxb = (unsigned short*)(w + 100663296);  // 16 MB, reuses qf

  // 1) casts
  cast_f32_bf16<<<32768, 256, 0, stream>>>(hs, hsb, 8388608);
  cast_transpose<<<dim3(128, 128), 256, 0, stream>>>(Wq, WqT, 4096, 4096);
  cast_transpose<<<dim3(128, 32),  256, 0, stream>>>(Wk, WkT, 1024, 4096);
  cast_transpose<<<dim3(128, 32),  256, 0, stream>>>(Wv, WvT, 1024, 4096);
  cast_transpose<<<dim3(128, 128), 256, 0, stream>>>(Wo, WoT, 4096, 4096);

  // 2) QKV projections (bf16 WMMA GEMM)
  gemm_bf16<<<dim3(32, 16), 256, 0, stream>>>(hsb, WqT, qf, 2048, 4096, 4096);
  gemm_bf16<<<dim3(8, 16),  256, 0, stream>>>(hsb, WkT, kf, 2048, 1024, 4096);
  gemm_bf16<<<dim3(8, 16),  256, 0, stream>>>(hsb, WvT, vf, 2048, 1024, 4096);

  // 3) RoPE + cache writes
  rope_q<<<32768, 256, 0, stream>>>(qf, cosp, sinp, qb);
  rope_copy_kv<<<8192, 256, 0, stream>>>(kf, vf, cosp, sinp, kcache, vcache, kb, vb);

  // 4) flash attention -> ctx bf16
  flash_attn<<<dim3(16, 32), 256, 0, stream>>>(qb, kb, vb, ctxb);

  // 5) output projection -> attn_out fp32
  gemm_bf16<<<dim3(32, 16), 256, 0, stream>>>(ctxb, WoT, attn, 2048, 4096, 4096);
}
